// MomentumQueue_51221779972733
// MI455X (gfx1250) — matrix operand order repaired and checked by reading, not compiled, not run on hardware
//
#include <hip/hip_runtime.h>
#include <hip/hip_bf16.h>

typedef __attribute__((ext_vector_type(16))) _Float16 v16h;
typedef __attribute__((ext_vector_type(8)))  _Float16 v8h;
typedef __attribute__((ext_vector_type(8)))  float    v8f;

#define NEG_INF (-__builtin_inff())

// Problem constants (fixed by the reference)
#define NQ    8192      // queue_size
#define NB    256       // batch
#define NTOT  8448      // NQ + NB
#define DIM   512       // feature_dim
#define NCLS  65        // num_cls
#define KNN   10        // K_NEIGHBORS
#define NSPLIT 16       // column splits for the fused sim+topk kernel
#define TILES_PER_SPLIT 33   // (8448/16)/NSPLIT

// ---------------------------------------------------------------------------
// 1) Row-normalize concat(memory, x) and store as f16 (values in [-1,1]).
// ---------------------------------------------------------------------------
__global__ __launch_bounds__(128) void k_normalize(const float* __restrict__ x,
                                                   const float* __restrict__ mem,
                                                   _Float16* __restrict__ sn) {
  int n = blockIdx.x;
  int t = threadIdx.x;
  const float* src = (n < NQ) ? (mem + (size_t)n * DIM)
                              : (x + (size_t)(n - NQ) * DIM);
  float ss = 0.f;
  for (int k = t; k < DIM; k += 128) { float v = src[k]; ss += v * v; }
  __shared__ float red[128];
  red[t] = ss;
  __syncthreads();
  for (int s = 64; s > 0; s >>= 1) {
    if (t < s) red[t] += red[t + s];
    __syncthreads();
  }
  float rn = 1.0f / fmaxf(sqrtf(red[0]), 1e-12f);
  for (int k = t; k < DIM; k += 128)
    sn[(size_t)n * DIM + k] = (_Float16)(src[k] * rn);
}

// ---------------------------------------------------------------------------
// 2) Fused sim = sn*sn^T (WMMA f16 -> f32) + per-row running top-10.
//    One wave per block; block owns 16 rows and a 528-column strip.
//    A fragments (16 x 512) live in VGPRs for the whole kernel.
//    B panels (16 cols x 512 K = 16 KB, contiguous rows of sn) are streamed
//    into LDS with double-buffered GLOBAL_LOAD_ASYNC_TO_LDS_B128 copies
//    (ASYNCcnt-tracked), fully overlapping global traffic with the WMMA chain.
// ---------------------------------------------------------------------------
__global__ __launch_bounds__(32) void k_simtopk(const _Float16* __restrict__ sn,
                                                float* __restrict__ pval,
                                                int* __restrict__ pidx) {
  const int lane = threadIdx.x;
  const int rb = blockIdx.x / NSPLIT;      // row block 0..527
  const int sp = blockIdx.x % NSPLIT;      // column split 0..15
  const int r0 = rb * 16;
  const int m  = lane & 15;
  const int hi = lane >> 4;                // 0 for lanes 0-15, 1 for 16-31

  // Double-buffered B panel: [buf][col][k], 16 KB per buffer.
  __shared__ __align__(64) _Float16 panel[2][16 * DIM];
  __shared__ float tile[16][17];

  float tv[KNN]; int ti[KNN];
#pragma unroll
  for (int j = 0; j < KNN; ++j) { tv[j] = NEG_INF; ti[j] = 0; }

  // ---- Preload the 16 A fragments (16x32 f16 each) into registers.
  // A layout: lane l holds row M=l&15; elems 0..15 = K khalf..khalf+15,
  // khalf = (l<16 ? 0 : 16).
  const _Float16* aBase = sn + (size_t)(r0 + m) * DIM + (hi ? 16 : 0);
  v16h af[16];
#pragma unroll
  for (int kc = 0; kc < DIM / 32; ++kc) {
    const _Float16* pA = aBase + kc * 32;
    v8h alo = *(const v8h*)(pA);
    v8h ahi = *(const v8h*)(pA + 8);
#pragma unroll
    for (int i = 0; i < 8; ++i) { af[kc][i] = alo[i]; af[kc][i + 8] = ahi[i]; }
  }

  // ---- Async panel copy: 16 KB contiguous (rows c0..c0+15 of sn) -> LDS.
  // 32 x global_load_async_to_lds_b128: each moves 32 lanes x 16 B = 512 B.
  const char* snBytes = (const char*)sn;
  auto issue_panel = [&](int buf, int c0) {
    unsigned lds = (unsigned)(unsigned long long)(const void*)&panel[buf][0]
                   + (unsigned)(lane * 16);
    unsigned long long ga =
        (unsigned long long)(const void*)(snBytes + (size_t)c0 * (DIM * 2))
        + (unsigned long long)(lane * 16);
#pragma unroll
    for (int i = 0; i < 32; ++i) {
      asm volatile("global_load_async_to_lds_b128 %0, %1, off"
                   :: "v"(lds), "v"(ga) : "memory");
      lds += 512;
      ga  += 512;
    }
  };

  // Prologue: start copy of tile 0.
  issue_panel(0, (sp * TILES_PER_SPLIT + 0) * 16);

  for (int ct = 0; ct < TILES_PER_SPLIT; ++ct) {
    const int c0 = (sp * TILES_PER_SPLIT + ct) * 16;

    // Prefetch next panel, then wait for the current one to be fully in LDS.
    // Async loads complete in order, so ASYNCcnt <= 32 => previous 32 done.
    if (ct + 1 < TILES_PER_SPLIT) {
      issue_panel((ct + 1) & 1, (sp * TILES_PER_SPLIT + ct + 1) * 16);
      asm volatile("s_wait_asynccnt 0x20" ::: "memory");
    } else {
      asm volatile("s_wait_asynccnt 0x0" ::: "memory");
    }

    const int buf = ct & 1;
    // B layout: lane l holds col N=l&15; elems 0..15 = K kb..kb+15,
    // kb = (l<16 ? 0 : 16). Contiguous 32 B from the panel row.
    const _Float16* bRow = &panel[buf][(size_t)m * DIM + (hi ? 16 : 0)];

    v8f acc = {};
#pragma unroll
    for (int kc = 0; kc < DIM / 32; ++kc) {
      v16h b = *(const v16h*)(bRow + kc * 32);
      acc = __builtin_amdgcn_wmma_f32_16x16x32_f16(
          /*neg_a=*/false, af[kc], /*neg_b=*/false, b,
          /*c_mod=*/(short)0, acc, /*reuse_a=*/false, /*reuse_b=*/false);
    }

    // C/D layout: VGPR r, lanes 0-15 -> M=r, lanes 16-31 -> M=r+8, N=lane&15
#pragma unroll
    for (int r = 0; r < 8; ++r) tile[r + (hi ? 8 : 0)][m] = acc[r];
    __syncthreads();

    if (lane < 16) {
      const int grow = r0 + lane;
#pragma unroll
      for (int j2 = 0; j2 < 16; ++j2) {
        const int gcol = c0 + j2;
        if (gcol == grow) continue;        // diagonal -> excluded
        const float v = tile[lane][j2];
        if (v > tv[KNN - 1]) {
          tv[KNN - 1] = v; ti[KNN - 1] = gcol;
#pragma unroll
          for (int q = KNN - 1; q > 0; --q) {
            if (tv[q] > tv[q - 1]) {
              float tf = tv[q]; tv[q] = tv[q - 1]; tv[q - 1] = tf;
              int   tt = ti[q]; ti[q] = ti[q - 1]; ti[q - 1] = tt;
            }
          }
        }
      }
    }
    __syncthreads();
  }

  if (lane < 16) {
    const size_t base = ((size_t)(r0 + lane) * NSPLIT + sp) * KNN;
#pragma unroll
    for (int j = 0; j < KNN; ++j) { pval[base + j] = tv[j]; pidx[base + j] = ti[j]; }
  }
}

// ---------------------------------------------------------------------------
// 3) Merge NSPLIT partial top-10 lists per row into the final sorted top-10.
// ---------------------------------------------------------------------------
__global__ __launch_bounds__(256) void k_merge(const float* __restrict__ pval,
                                               const int* __restrict__ pidx,
                                               float* __restrict__ wout,
                                               int* __restrict__ iout) {
  const int n = blockIdx.x * blockDim.x + threadIdx.x;
  if (n >= NTOT) return;
  float tv[KNN]; int ti[KNN];
#pragma unroll
  for (int j = 0; j < KNN; ++j) { tv[j] = NEG_INF; ti[j] = 0; }
  const size_t base = (size_t)n * NSPLIT * KNN;
  for (int s = 0; s < NSPLIT * KNN; ++s) {
    const float v = pval[base + s];
    if (v > tv[KNN - 1]) {
      tv[KNN - 1] = v; ti[KNN - 1] = pidx[base + s];
#pragma unroll
      for (int q = KNN - 1; q > 0; --q) {
        if (tv[q] > tv[q - 1]) {
          float tf = tv[q]; tv[q] = tv[q - 1]; tv[q - 1] = tf;
          int   tt = ti[q]; ti[q] = ti[q - 1]; ti[q - 1] = tt;
        }
      }
    }
  }
#pragma unroll
  for (int j = 0; j < KNN; ++j) {
    wout[(size_t)n * KNN + j] = tv[j];
    iout[(size_t)n * KNN + j] = ti[j];
  }
}

// ---------------------------------------------------------------------------
// 4) h0 = [one_hot(memory_label); zeros(batch)]
// ---------------------------------------------------------------------------
__global__ __launch_bounds__(256) void k_inith(const long long* __restrict__ label,
                                               float* __restrict__ h) {
  const int i = blockIdx.x * blockDim.x + threadIdx.x;
  if (i >= NTOT * NCLS) return;
  const int n = i / NCLS, c = i % NCLS;
  float v = 0.f;
  if (n < NQ) v = (label[n] == (long long)c) ? 1.0f : 0.0f;
  h[i] = v;
}

// ---------------------------------------------------------------------------
// 5) One GNN pass: h_out = softmax( (sum_k w[n,k]*h_in[idx[n,k]]) @ W )
// ---------------------------------------------------------------------------
__global__ __launch_bounds__(128) void k_gnn(const float* __restrict__ hin,
                                             float* __restrict__ hout,
                                             const float* __restrict__ wG,
                                             const int* __restrict__ idxG,
                                             const float* __restrict__ Wm) {
  const int n = blockIdx.x;
  const int t = threadIdx.x;
  __shared__ float sAgg[NCLS];
  __shared__ float sO[NCLS];

  if (t < NCLS) {
    float acc = 0.f;
#pragma unroll
    for (int k = 0; k < KNN; ++k) {
      const float wk = wG[(size_t)n * KNN + k];
      const int   j  = idxG[(size_t)n * KNN + k];
      acc += wk * hin[(size_t)j * NCLS + t];
    }
    sAgg[t] = acc;
  }
  __syncthreads();
  if (t < NCLS) {
    float o = 0.f;
    for (int c = 0; c < NCLS; ++c) o += sAgg[c] * Wm[c * NCLS + t];
    sO[t] = o;
  }
  __syncthreads();
  float e = 0.f;
  if (t < NCLS) {
    float mx = NEG_INF;
    for (int c = 0; c < NCLS; ++c) mx = fmaxf(mx, sO[c]);
    e = __expf(sO[t] - mx);
  }
  __syncthreads();
  if (t < NCLS) sAgg[t] = e;
  __syncthreads();
  if (t < NCLS) {
    float sum = 0.f;
    for (int c = 0; c < NCLS; ++c) sum += sAgg[c];
    hout[(size_t)n * NCLS + t] = e / sum;
  }
}

// ---------------------------------------------------------------------------
// 6) confidence = max(probs), predict = argmax(probs) over last 256 rows.
// ---------------------------------------------------------------------------
__global__ __launch_bounds__(256) void k_final(const float* __restrict__ h,
                                               float* __restrict__ out) {
  const int i = blockIdx.x * blockDim.x + threadIdx.x;
  if (i >= NB) return;
  const float* row = h + (size_t)(NQ + i) * NCLS;
  float best = row[0]; int arg = 0;
  for (int c = 1; c < NCLS; ++c) {
    const float v = row[c];
    if (v > best) { best = v; arg = c; }
  }
  out[i]      = best;
  out[NB + i] = (float)arg;
}

// ---------------------------------------------------------------------------
extern "C" void kernel_launch(void* const* d_in, const int* in_sizes, int n_in,
                              void* d_out, int out_size, void* d_ws, size_t ws_size,
                              hipStream_t stream) {
  (void)in_sizes; (void)n_in; (void)out_size; (void)ws_size;
  const float*     x     = (const float*)d_in[0];      // (256, 512)
  const float*     mem   = (const float*)d_in[1];      // (8192, 512)
  const float*     Wm    = (const float*)d_in[2];      // (65, 65)
  const long long* label = (const long long*)d_in[3];  // (8192,) int64
  float* out = (float*)d_out;                          // [conf(256) | predict(256)]

  // Workspace carve-out (256B aligned)
  char* ws = (char*)d_ws;
  size_t off = 0;
  auto carve = [&](size_t bytes) -> char* {
    char* p = ws + off;
    off += (bytes + 255) & ~(size_t)255;
    return p;
  };
  _Float16* sn   = (_Float16*)carve((size_t)NTOT * DIM * sizeof(_Float16));   // 8.65 MB
  float*    pval = (float*)   carve((size_t)NTOT * NSPLIT * KNN * sizeof(float));
  int*      pidx = (int*)     carve((size_t)NTOT * NSPLIT * KNN * sizeof(int));
  float*    wfin = (float*)   carve((size_t)NTOT * KNN * sizeof(float));
  int*      ifin = (int*)     carve((size_t)NTOT * KNN * sizeof(int));
  float*    hA   = (float*)   carve((size_t)NTOT * NCLS * sizeof(float));
  float*    hB   = (float*)   carve((size_t)NTOT * NCLS * sizeof(float));

  // 1) normalize + f16 cast
  k_normalize<<<NTOT, 128, 0, stream>>>(x, mem, sn);

  // 2) fused WMMA similarity + per-split top-10 (8448 single-wave blocks)
  k_simtopk<<<528 * NSPLIT, 32, 0, stream>>>(sn, pval, pidx);

  // 3) merge splits -> final sorted top-10 weights/indices
  k_merge<<<(NTOT + 255) / 256, 256, 0, stream>>>(pval, pidx, wfin, ifin);

  // 4) h0 = [one_hot(labels); zeros]
  k_inith<<<(NTOT * NCLS + 255) / 256, 256, 0, stream>>>(label, hA);

  // 5) three GNN passes (A->B->A->B)
  k_gnn<<<NTOT, 128, 0, stream>>>(hA, hB, wfin, ifin, Wm);
  k_gnn<<<NTOT, 128, 0, stream>>>(hB, hA, wfin, ifin, Wm);
  k_gnn<<<NTOT, 128, 0, stream>>>(hA, hB, wfin, ifin, Wm);

  // 6) confidence + predict for the batch rows
  k_final<<<1, 256, 0, stream>>>(hB, out);
}
